// LSTM_23330262352184
// MI455X (gfx1250) — compile-verified, matrix operand order checked by reading, not run on hardware
//
#include <hip/hip_runtime.h>

typedef __attribute__((ext_vector_type(16))) __bf16 v16bf;
typedef __attribute__((ext_vector_type(8)))  float  v8f;

#define BB   256
#define TT   128
#define DIN  256
#define EE   512
#define HH   512
#define FH   2048            // 4*H
#define ROWS (BB * TT)       // 32768
#define RBLOCKS 64           // persistent recurrence blocks (512 waves / 8)
#define WH_LDS_BYTES (4 * 16 * 32 * 32)   // 4 gates x 16 k-chunks x 32 lanes x 32B = 64KB

// ---------------------------------------------------------------------------
// elementwise f32 -> bf16 convert
__global__ void LSTM_cvt_bf16(const float* __restrict__ src,
                              __bf16* __restrict__ dst, int n) {
    int i = blockIdx.x * blockDim.x + threadIdx.x;
    if (i < n) dst[i] = (__bf16)src[i];
}

// transpose-convert: src [K,N] f32 row-major  ->  dst [N,K] bf16 row-major
__global__ void LSTM_transpose_bf16(const float* __restrict__ src,
                                    __bf16* __restrict__ dst, int K, int N) {
    int i = blockIdx.x * blockDim.x + threadIdx.x;
    if (i >= K * N) return;
    int n = i / K;
    int k = i - n * K;
    dst[i] = (__bf16)src[(size_t)k * N + n];
}

// zero h0 (bf16), the prepended zero timestep of the output, and the
// grid-barrier counters (one per timestep; re-zeroed every call).
__global__ void LSTM_init(__bf16* __restrict__ h0,
                          float* __restrict__ stacked,
                          unsigned* __restrict__ barrier_ctr) {
    int i = blockIdx.x * blockDim.x + threadIdx.x;  // 0 .. B*H-1
    if (i >= BB * HH) return;
    int b = i / HH, h = i - b * HH;
    h0[i] = (__bf16)0.0f;
    stacked[(size_t)b * (TT + 1) * HH + h] = 0.0f;
    if (i < TT) barrier_ctr[i] = 0u;
}

// ---------------------------------------------------------------------------
// Generic WMMA GEMM:  out = act(A[M,K]bf16 @ Bt[N,K]^T bf16 + bias)
// Each wave computes NT adjacent 16x16 tiles, reusing the A fragment.
// Register double-buffered along K so loads for chunk k+1 are in flight
// while the WMMAs for chunk k execute (partial s_wait_loadcnt).
template <int K, int NT, bool RELU, bool OUT_BF16>
__global__ void LSTM_gemm_wmma(const __bf16* __restrict__ A,
                               const __bf16* __restrict__ Bt,
                               const float* __restrict__ bias,
                               __bf16* __restrict__ outb,
                               float* __restrict__ outf,
                               int Mrows, int N) {
    const int lane = threadIdx.x & 31;
    const int wave = blockIdx.x * (blockDim.x >> 5) + (threadIdx.x >> 5);
    const int ngroups = N / (16 * NT);
    const int mt = wave / ngroups;
    const int ng = wave - mt * ngroups;
    if (mt * 16 >= Mrows) return;                 // uniform per wave
    const int m0   = mt * 16;
    const int n0   = ng * 16 * NT;
    const int half = lane >> 4;                   // 0: K 0..15, 1: K 16..31
    const int l15  = lane & 15;

    v8f acc[NT];
#pragma unroll
    for (int j = 0; j < NT; ++j) acc[j] = (v8f){0, 0, 0, 0, 0, 0, 0, 0};

    const __bf16* aB = A + (size_t)(m0 + l15) * K + half * 16;
    const __bf16* bB[NT];
#pragma unroll
    for (int j = 0; j < NT; ++j)
        bB[j] = Bt + (size_t)(n0 + j * 16 + l15) * K + half * 16;

    v16bf a_cur = *(const v16bf*)aB;
    v16bf b_cur[NT];
#pragma unroll
    for (int j = 0; j < NT; ++j) b_cur[j] = *(const v16bf*)bB[j];

#pragma unroll
    for (int kt = 0; kt < K; kt += 32) {
        v16bf a_nxt = a_cur;
        v16bf b_nxt[NT];
#pragma unroll
        for (int j = 0; j < NT; ++j) b_nxt[j] = b_cur[j];
        if (kt + 32 < K) {                         // compile-time resolved
            a_nxt = *(const v16bf*)(aB + kt + 32);
#pragma unroll
            for (int j = 0; j < NT; ++j)
                b_nxt[j] = *(const v16bf*)(bB[j] + kt + 32);
        }
#pragma unroll
        for (int j = 0; j < NT; ++j)
            acc[j] = __builtin_amdgcn_wmma_f32_16x16x32_bf16(
                false, a_cur, false, b_cur[j], (short)0, acc[j], false, false);
        a_cur = a_nxt;
#pragma unroll
        for (int j = 0; j < NT; ++j) b_cur[j] = b_nxt[j];
    }

    // C/D layout: VGPR r -> M = m0 + r + 8*half,  N = n0 + j*16 + (lane&15)
#pragma unroll
    for (int j = 0; j < NT; ++j) {
        const int n = n0 + j * 16 + l15;
        const float bv = bias ? bias[n] : 0.0f;
#pragma unroll
        for (int r = 0; r < 8; ++r) {
            const int m = m0 + r + half * 8;
            float v = acc[j][r] + bv;
            if (RELU) v = v > 0.0f ? v : 0.0f;
            if (OUT_BF16) outb[(size_t)m * N + n] = (__bf16)v;
            else          outf[(size_t)m * N + n] = v;
        }
    }
}

// ---------------------------------------------------------------------------
__device__ __forceinline__ float sigmoidf_(float x) {
    return 1.0f / (1.0f + __expf(-x));
}

// Software grid barrier: one counter per timestep (zeroed by LSTM_init each
// call). Agent-scope acq-rel publishes this block's h stores to all blocks.
__device__ __forceinline__ void LSTM_grid_barrier(unsigned* __restrict__ ctr,
                                                  int t) {
    __syncthreads();
    if (threadIdx.x == 0) {
        __hip_atomic_fetch_add(&ctr[t], 1u, __ATOMIC_ACQ_REL,
                               __HIP_MEMORY_SCOPE_AGENT);
        while (__hip_atomic_load(&ctr[t], __ATOMIC_ACQUIRE,
                                 __HIP_MEMORY_SCOPE_AGENT) < (unsigned)RBLOCKS)
            __builtin_amdgcn_s_sleep(1);
    }
    __syncthreads();
}

// Persistent LSTM recurrence: ONE launch runs all T timesteps.
// 64 blocks x 8 waves = 512 waves. All 8 waves of a block share one n-tile
// (nt = blockIdx&31) and cover 8 m-tiles, so the block's recurrent weights
// (4 gate tiles x 16 cols x 512 K = 64KB bf16) are staged in LDS ONCE and
// re-read per step with ds_load_b128 (opaque index defeats LICM/spilling).
// Cell state c lives in 8 VGPRs across all 128 steps. h ping-pongs through
// two global bf16 buffers published via the grid barrier.
__global__ void LSTM_recurrence(__bf16* __restrict__ hb0,
                                __bf16* __restrict__ hb1,
                                const __bf16* __restrict__ WhT,  // [4H,H] bf16
                                const float* __restrict__ zi,    // [B,T,4H]
                                const float* __restrict__ bh,    // [4H]
                                float* __restrict__ stacked,     // [B,T+1,H]
                                float* __restrict__ cdst,        // [B,H] tail
                                float* __restrict__ hdst,        // [B,H] tail
                                unsigned* __restrict__ barrier_ctr) {
    extern __shared__ __align__(32) __bf16 shB[];   // 64KB: (g,kchunk,lane)->32B

    const int lane = threadIdx.x & 31;
    const int wv   = threadIdx.x >> 5;           // 0..7
    const int nt   = blockIdx.x & 31;            // shared by all waves in block
    const int mt   = (blockIdx.x >> 5) * 8 + wv; // 0..15
    const int m0   = mt * 16;
    const int n0   = nt * 16;
    const int half = lane >> 4;
    const int l15  = lane & 15;
    const int n    = n0 + l15;

    // ---- stage this block's recurrent weights into LDS (once) -------------
    // chunk c = (g*16 + kchunk)*32 + lane'  ->  32 bytes
    for (int c = threadIdx.x; c < 4 * 16 * 32; c += 256) {
        const int g  = c >> 9;
        const int rm = c & 511;
        const int kc = rm >> 5;
        const int lp = rm & 31;
        const __bf16* src = WhT + (size_t)(g * HH + n0 + (lp & 15)) * HH +
                            kc * 32 + (lp >> 4) * 16;
        *(v16bf*)(shB + (size_t)c * 16) = *(const v16bf*)src;
    }
    __syncthreads();

    const float bi = bh[n],          bf_ = bh[n + HH],
                bg = bh[n + 2 * HH], bo  = bh[n + 3 * HH];

    float creg[8];
#pragma unroll
    for (int r = 0; r < 8; ++r) creg[r] = 0.0f;

    unsigned inv = 0;        // runtime 0; made opaque per-iteration below

#pragma unroll 1
    for (int t = 0; t < TT; ++t) {
        // Opaque index: compiler cannot prove LDS fragments loop-invariant,
        // so it must ds_load_b128 per step instead of hoisting + spilling.
        asm volatile("" : "+v"(inv));
        const __bf16* shW = shB + inv;

        const __bf16* hp = (t & 1) ? hb1 : hb0;
        __bf16*       hn = (t & 1) ? hb0 : hb1;

        v8f acc[4];
#pragma unroll
        for (int g = 0; g < 4; ++g) acc[g] = (v8f){0, 0, 0, 0, 0, 0, 0, 0};

        const __bf16* aB = hp + (size_t)(m0 + l15) * HH + half * 16;
        v16bf a_cur = *(const v16bf*)aB;

#pragma unroll
        for (int kc = 0; kc < 16; ++kc) {
            v16bf a_nxt = a_cur;
            if (kc + 1 < 16)
                a_nxt = *(const v16bf*)(aB + (kc + 1) * 32);
#pragma unroll
            for (int g = 0; g < 4; ++g) {
                v16bf b = *(const v16bf*)(shW +
                    (size_t)(((g * 16 + kc) * 32 + lane) * 16));
                acc[g] = __builtin_amdgcn_wmma_f32_16x16x32_bf16(
                    false, a_cur, false, b, (short)0, acc[g], false, false);
            }
            a_cur = a_nxt;
        }

        // gates + state update; c stays in registers
#pragma unroll
        for (int r = 0; r < 8; ++r) {
            const int b = m0 + r + half * 8;
            const size_t zoff = ((size_t)b * TT + t) * FH;
            const float zii = zi[zoff + n]          + acc[0][r] + bi;
            const float zif = zi[zoff + n + HH]     + acc[1][r] + bf_;
            const float zig = zi[zoff + n + 2 * HH] + acc[2][r] + bg;
            const float zio = zi[zoff + n + 3 * HH] + acc[3][r] + bo;
            const float ig = sigmoidf_(zii);
            const float fg = sigmoidf_(zif);
            const float gg = tanhf(zig);
            const float og = sigmoidf_(zio);
            const float cn = fg * creg[r] + ig * gg;
            creg[r] = cn;
            const float hv = og * tanhf(cn);
            hn[(size_t)b * HH + n] = (__bf16)hv;
            stacked[((size_t)b * (TT + 1) + (t + 1)) * HH + n] = hv;
            if (t == TT - 1) {
                cdst[(size_t)b * HH + n] = cn;
                hdst[(size_t)b * HH + n] = hv;
            }
        }

        if (t != TT - 1) LSTM_grid_barrier(barrier_ctr, t);
    }
}

// ---------------------------------------------------------------------------
extern "C" void kernel_launch(void* const* d_in, const int* in_sizes, int n_in,
                              void* d_out, int out_size, void* d_ws,
                              size_t ws_size, hipStream_t stream) {
    const float* seq = (const float*)d_in[0];
    const float* W1  = (const float*)d_in[1];
    const float* b1  = (const float*)d_in[2];
    const float* W2  = (const float*)d_in[3];
    const float* b2  = (const float*)d_in[4];
    const float* Wi  = (const float*)d_in[5];
    const float* Wh  = (const float*)d_in[6];
    const float* bh  = (const float*)d_in[7];
    float* out = (float*)d_out;

    // workspace layout (256B aligned slabs)
    char* base = (char*)d_ws;
    size_t off = 0;
    auto alloc = [&](size_t bytes) {
        char* p = base + off;
        off = (off + bytes + 255) & ~(size_t)255;
        return p;
    };
    __bf16*   seqb    = (__bf16*)alloc((size_t)ROWS * DIN * 2);   // 16 MB
    __bf16*   w1t     = (__bf16*)alloc((size_t)512 * DIN * 2);
    __bf16*   w2t     = (__bf16*)alloc((size_t)EE * 512 * 2);
    __bf16*   wit     = (__bf16*)alloc((size_t)FH * EE * 2);
    __bf16*   wht     = (__bf16*)alloc((size_t)FH * HH * 2);
    __bf16*   hiddenb = (__bf16*)alloc((size_t)ROWS * 512 * 2);   // 33.5 MB
    __bf16*   embb    = (__bf16*)alloc((size_t)ROWS * EE * 2);    // 33.5 MB
    float*    zi      = (float*) alloc((size_t)ROWS * FH * 4);    // 268 MB
    __bf16*   hbuf0   = (__bf16*)alloc((size_t)BB * HH * 2);
    __bf16*   hbuf1   = (__bf16*)alloc((size_t)BB * HH * 2);
    unsigned* bctr    = (unsigned*)alloc((size_t)TT * 4);

    // --- precision prep: f32 -> bf16, weights pre-transposed to [N,K] ------
    {
        int n = ROWS * DIN;
        LSTM_cvt_bf16<<<(n + 255) / 256, 256, 0, stream>>>(seq, seqb, n);
    }
    LSTM_transpose_bf16<<<(DIN * 512 + 255) / 256, 256, 0, stream>>>(W1, w1t, DIN, 512);
    LSTM_transpose_bf16<<<(512 * EE  + 255) / 256, 256, 0, stream>>>(W2, w2t, 512, EE);
    LSTM_transpose_bf16<<<(EE  * FH  + 255) / 256, 256, 0, stream>>>(Wi, wit, EE, FH);
    LSTM_transpose_bf16<<<(HH  * FH  + 255) / 256, 256, 0, stream>>>(Wh, wht, HH, FH);
    LSTM_init<<<(BB * HH + 255) / 256, 256, 0, stream>>>(hbuf0, out, bctr);

    // --- embed MLP + input projection, all WMMA ----------------------------
    // hidden = relu(seq @ W1 + b1)        [32768,256]x[256,512] -> bf16
    {
        int waves = (ROWS / 16) * (512 / 64);
        LSTM_gemm_wmma<DIN, 4, true, true><<<waves / 8, 256, 0, stream>>>(
            seqb, w1t, b1, hiddenb, nullptr, ROWS, 512);
    }
    // emb = hidden @ W2 + b2              [32768,512]x[512,512] -> bf16
    {
        int waves = (ROWS / 16) * (EE / 64);
        LSTM_gemm_wmma<512, 4, false, true><<<waves / 8, 256, 0, stream>>>(
            hiddenb, w2t, b2, embb, nullptr, ROWS, EE);
    }
    // zi_all = emb @ Wi                   [32768,512]x[512,2048] -> f32
    {
        int waves = (ROWS / 16) * (FH / 64);
        LSTM_gemm_wmma<EE, 4, false, false><<<waves / 8, 256, 0, stream>>>(
            embb, wit, nullptr, nullptr, zi, ROWS, FH);
    }

    // --- LSTM recurrence: ONE persistent WMMA kernel, LDS-resident Wh,
    //     software grid-barrier per step ----------------------------------
    {
        float* cdst = out + (size_t)BB * (TT + 1) * HH;
        float* hdst = cdst + (size_t)BB * HH;
        LSTM_recurrence<<<RBLOCKS, 256, WH_LDS_BYTES, stream>>>(
            hbuf0, hbuf1, wht, zi, bh, out, cdst, hdst, bctr);
    }
    (void)in_sizes; (void)n_in; (void)out_size; (void)ws_size;
}
